// get_model_79860621902607
// MI455X (gfx1250) — matrix operand order repaired and checked by reading, not compiled, not run on hardware
//
#include <hip/hip_runtime.h>
#include <stdint.h>

// ---------------- CDNA5 WMMA plumbing (bf16 16x16x32, f32 accum) ----------------
typedef __attribute__((ext_vector_type(16))) __bf16 v16bf;
typedef __attribute__((ext_vector_type(8)))  float  v8f;

#define DEVINL __device__ __forceinline__

DEVINL uint16_t f2bf(float x) {               // round-to-nearest-even f32 -> bf16
  uint32_t u = __float_as_uint(x);
  uint32_t r = u + 0x7FFFu + ((u >> 16) & 1u);
  return (uint16_t)(r >> 16);
}

union FragU { uint32_t u[8]; v16bf v; };

// ---- fragment-major storage ----
// All WMMA operands are stored "fragment-major": for tile (mt|nt, ks) each lane's
// 8 dwords are contiguous (32B), so a fragment load is two b128 loads.
// half index of A-operand element (m, c) for an [M x C] matrix, KS = C/32:
DEVINL int frag_half_index(int m, int c, int KS) {
  int mt = m >> 4, mr = m & 15;
  int ks = c >> 5, k32 = c & 31;
  int grp = k32 >> 3;                              // 0..3
  int lane = mr | ((grp & 1) << 4);                // hi-lane holds k32 8-15 / 24-31
  int slot = ((grp >> 1) << 2) | ((k32 & 7) >> 1); // dword slot 0..7
  return (((mt * KS + ks) * 32 + lane) << 4) | (slot << 1) | (c & 1);
}

// contiguous fragment load: lane's 8 dwords at ((tile*KS + ks)*32 + lane)*8
DEVINL v16bf load_frag_sw(const uint16_t* base, int KS, int tile, int ks, int lane) {
  const uint32_t* p = (const uint32_t*)base + ((((size_t)tile * KS + ks) * 32 + lane) << 3);
  FragU f;
#pragma unroll
  for (int i = 0; i < 8; ++i) f.u[i] = p[i];       // merges into 2x b128
  return f.v;
}

DEVINL v8f wmma_bf16(v16bf a, v16bf b, v8f c) {
  return __builtin_amdgcn_wmma_f32_16x16x32_bf16(false, a, false, b, (short)0, c, false, false);
}

// One fused MLP layer over a fragment-major [KP x CINP] LDS tile -> [KP x NP] (relu).
// OUT_BF: write result directly into next layer's fragment-major A layout (K-dim NP).
template<int KP, int CINP, int NP, bool OUT_BF>
DEVINL void wmma_layer(const uint16_t* ldsIn, const uint32_t* Wf, const float* bias,
                       uint16_t* outBf, float* outF, int wave, int lane) {
  constexpr int MT = KP / 16, NT = NP / 16, KS = CINP / 32, KS2 = NP / 32;
  for (int t = wave; t < MT * NT; t += 4) {
    int mt = t % MT, nt = t / MT;
    v8f acc = {};
    for (int ks = 0; ks < KS; ++ks) {
      v16bf af = load_frag_sw(ldsIn, KS, mt, ks, lane);
      v16bf bf = load_frag_sw((const uint16_t*)Wf, KS, nt, ks, lane);
      acc = wmma_bf16(af, bf, acc);
    }
    int n = nt * 16 + (lane & 15);
    float bi = bias[n];
    int mofs = mt * 16 + ((lane >> 4) << 3);   // D: VGPR r -> M = r (+8 for hi lanes)
#pragma unroll
    for (int r = 0; r < 8; ++r) {
      float v = acc[r] + bi;
      v = v > 0.f ? v : 0.f;
      if (OUT_BF) outBf[frag_half_index(mofs + r, n, KS2)] = f2bf(v);
      else        outF [(mofs + r) * NP + n] = v;
    }
  }
}

// ---------------- misc small kernels ----------------
__global__ void transpose_kernel(const float* __restrict__ xyz, float* __restrict__ pts, int N) {
  int t = blockIdx.x * blockDim.x + threadIdx.x;           // pts[b][n][c]
  if (t >= 32 * N * 3) return;
  int b = t / (N * 3); int r = t % (N * 3); int n = r / 3; int c = r % 3;
  pts[t] = xyz[((size_t)b * 3 + c) * N + n];
}

// Fold BN into weights, pad, convert to bf16, and store B fragment-major:
// B element (k, n): lane = (n&15) | (((k32>>3)&1)<<4), slot = ((k32>>4)<<2) | ((k32&7)>>1), tile nt = n>>4.
__global__ void prep_w_kernel(const float* __restrict__ w, const float* __restrict__ b,
                              const float* __restrict__ g, const float* __restrict__ be,
                              uint32_t* __restrict__ Wf, float* __restrict__ biasOut,
                              int Cin, int Cout, int CinP, int CoutP) {
  int pairs = (CinP >> 1) * CoutP;
  int KS = CinP >> 5;
  int t = blockIdx.x * blockDim.x + threadIdx.x;
  if (t < pairs) {
    int kp = t / CoutP, n = t % CoutP;
    int k0 = kp * 2, k1 = k0 + 1;
    float gn = (g && n < Cout) ? g[n] : 1.f;
    float v0 = (k0 < Cin && n < Cout) ? w[(size_t)k0 * Cout + n] * gn : 0.f;
    float v1 = (k1 < Cin && n < Cout) ? w[(size_t)k1 * Cout + n] * gn : 0.f;
    int ks = k0 >> 5, k32 = k0 & 31;
    int lane = (n & 15) | (((k32 >> 3) & 1) << 4);
    int slot = ((k32 >> 4) << 2) | ((k32 & 7) >> 1);
    int nt = n >> 4;
    Wf[(((size_t)nt * KS + ks) * 32 + lane) * 8 + slot] =
        (uint32_t)f2bf(v0) | ((uint32_t)f2bf(v1) << 16);
  } else if (t < pairs + CoutP) {
    int n = t - pairs;
    float bb = 0.f;
    if (n < Cout) { bb = b[n]; if (g) bb = bb * g[n] + be[n]; }
    biasOut[n] = bb;
  }
}

__global__ void gather_xyz_kernel(const float* __restrict__ pts, const int* __restrict__ cent,
                                  float* __restrict__ out, int N, int S) {
  int t = blockIdx.x * blockDim.x + threadIdx.x;
  if (t >= 32 * S * 3) return;
  int c = t % 3; int g = t / 3; int b = g / S;
  out[t] = pts[((size_t)b * N + cent[g]) * 3 + c];
}

__global__ void zero_u32_kernel(unsigned* __restrict__ p, int n) {
  int t = blockIdx.x * blockDim.x + threadIdx.x;
  if (t < n) p[t] = 0u;
}

// uint-bits(f32) -> bf16, written fragment-major for the fc1 A operand (M=32, K=1024)
__global__ void u2bf_kernel(const unsigned* __restrict__ in, uint16_t* __restrict__ out, int n) {
  int t = blockIdx.x * blockDim.x + threadIdx.x;
  if (t < n) out[frag_half_index(t >> 10, t & 1023, 32)] = f2bf(__uint_as_float(in[t]));
}

// A3[b] is [128 x 544] fragment-major (MT=8, KS=17)
__global__ void build_a3_kernel(const float* __restrict__ l2, const float* __restrict__ l2xyz,
                                uint16_t* __restrict__ A3) {
  int t = blockIdx.x * blockDim.x + threadIdx.x;
  if (t >= 32 * 128 * 544) return;
  int c = t % 544; int rk = t / 544;
  int b = rk >> 7, k = rk & 127;
  float v = 0.f;
  if (c < 512)      v = l2[(size_t)rk * 512 + c];
  else if (c < 515) v = l2xyz[(size_t)rk * 3 + (c - 512)];
  A3[(size_t)b * 128 * 544 + frag_half_index(k, c, 17)] = f2bf(v);
}

__global__ void finalize_kernel(const float* __restrict__ fc3out, float* __restrict__ out) {
  int t = blockIdx.x * blockDim.x + threadIdx.x;
  if (t < 1280)      { int i = t / 40, j = t % 40; out[t] = fc3out[i * 48 + j]; }
  else if (t == 1280) out[t] = 0.f;      // cos_loss
  else if (t == 1281) out[t] = 512.f;
  else if (t == 1282) out[t] = 128.f;
}

// ---------------- FPS: one block per batch, cloud resident in LDS ----------------
template<int NMAX>
__global__ __launch_bounds__(256)
void fps_kernel(const float* __restrict__ pts, int* __restrict__ centers, int N, int npoint) {
  __shared__ float sx[NMAX], sy[NMAX], sz[NMAX], dist[NMAX];
  __shared__ float rv[256]; __shared__ int ri[256];
  int b = blockIdx.x, tid = threadIdx.x;
  for (int p = tid; p < N; p += 256) {
    const float* pp = pts + ((size_t)b * N + p) * 3;
    sx[p] = pp[0]; sy[p] = pp[1]; sz[p] = pp[2];
    dist[p] = 1e10f;
  }
  __syncthreads();
  int far = 0;
  for (int it = 0; it < npoint; ++it) {
    if (tid == 0) centers[(size_t)b * npoint + it] = far;   // record, then update (matches scan)
    float fx = sx[far], fy = sy[far], fz = sz[far];
    float bv = -1.f; int bi = 0;
    for (int p = tid; p < N; p += 256) {
      float dx = sx[p] - fx, dy = sy[p] - fy, dz = sz[p] - fz;
      float nd = fminf(dist[p], dx * dx + dy * dy + dz * dz);
      dist[p] = nd;
      if (nd > bv) { bv = nd; bi = p; }
    }
    rv[tid] = bv; ri[tid] = bi;
    __syncthreads();
    for (int off = 128; off > 0; off >>= 1) {
      if (tid < off) {
        float v2 = rv[tid + off]; int i2 = ri[tid + off];
        if (v2 > rv[tid] || (v2 == rv[tid] && i2 < ri[tid])) { rv[tid] = v2; ri[tid] = i2; }
      }
      __syncthreads();
    }
    far = ri[0];
    __syncthreads();
  }
}

// ---------------- ball query: one wave32 per group, ballot-ordered first-K ----------------
__global__ __launch_bounds__(128)
void ball_query_kernel(const float* __restrict__ pts, const int* __restrict__ centers,
                       int* __restrict__ gidx, int N, int S, int BS, float r2, int nsample) {
  int wave = threadIdx.x >> 5, lane = threadIdx.x & 31;
  int g = blockIdx.x * 4 + wave;
  if (g >= BS) return;
  int b = g / S, s = g % S;
  int cIdx = centers ? centers[g] : s;
  const float* cp = pts + ((size_t)b * N + cIdx) * 3;
  float cx = cp[0], cy = cp[1], cz = cp[2];
  int* out = gidx + (size_t)g * nsample;
  int cnt = 0, first = 0; bool haveFirst = false;
  for (int base = 0; base < N && cnt < nsample; base += 32) {
    int p = base + lane;
    bool in = false;
    if (p < N) {
      const float* pp = pts + ((size_t)b * N + p) * 3;
      float dx = pp[0] - cx, dy = pp[1] - cy, dz = pp[2] - cz;
      in = (dx * dx + dy * dy + dz * dz) <= r2;
    }
    unsigned m = (unsigned)__ballot(in);
    if (!haveFirst && m) { first = base + __builtin_ctz(m); haveFirst = true; }
    int pos = __builtin_popcount(m & ((1u << lane) - 1u));
    if (in && cnt + pos < nsample) out[cnt + pos] = p;
    cnt += __builtin_popcount(m);
  }
  int filled = cnt < nsample ? cnt : nsample;
  for (int t = lane; t < nsample; t += 32)
    if (t >= filled) out[t] = first;
}

// -------- fused: gather group -> MLP(2 layers, WMMA) -> max||mean pool, all in LDS --------
template<int KSAMP, int KP, int CF, int CINP, int C1P, int C2P, bool HASF>
__global__ __launch_bounds__(128)
void group_mlp2_pool_kernel(const float* __restrict__ pts, const float* __restrict__ feats,
                            const int* __restrict__ centers, const int* __restrict__ gidx,
                            const uint32_t* __restrict__ W1, const float* __restrict__ b1,
                            const uint32_t* __restrict__ W2, const float* __restrict__ b2,
                            float* __restrict__ out, int N, int S) {
  __shared__ alignas(16) uint16_t ldsA [KP * CINP];   // fragment-major
  __shared__ alignas(16) uint16_t ldsH1[KP * C1P];    // fragment-major
  __shared__ float                ldsH2[KP * C2P];    // linear, for pooling
  int b = blockIdx.x / S, s = blockIdx.x % S;
  int wave = threadIdx.x >> 5, lane = threadIdx.x & 31;
  int cIdx = centers ? centers[b * S + s] : s;
  const float* cp = pts + ((size_t)b * N + cIdx) * 3;
  float cx = cp[0], cy = cp[1], cz = cp[2];
  const int* gi = gidx + ((size_t)b * S + s) * KSAMP;
  for (int e = threadIdx.x; e < KP * CINP; e += 128) {
    int k = e / CINP, c = e % CINP;
    float v = 0.f;
    if (k < KSAMP) {
      int j = gi[k];
      if (HASF && c < CF) v = feats[((size_t)b * N + j) * CF + c];
      else if (c < CF + 3) {
        int d = c - CF;
        v = pts[((size_t)b * N + j) * 3 + d] - (d == 0 ? cx : (d == 1 ? cy : cz));
      } else if (c < CF + 6) {
        int d = c - CF - 3;
        v = (d == 0 ? cx : (d == 1 ? cy : cz));
      }
    }
    ldsA[frag_half_index(k, c, CINP / 32)] = f2bf(v);
  }
  __syncthreads();
  wmma_layer<KP, CINP, C1P, true >(ldsA,  W1, b1, ldsH1, (float*)nullptr, wave, lane);
  __syncthreads();
  wmma_layer<KP, C1P,  C2P, false>(ldsH1, W2, b2, (uint16_t*)nullptr, ldsH2, wave, lane);
  __syncthreads();
  float inv = 1.f / (float)KSAMP;
  float* o = out + ((size_t)b * S + s) * (size_t)(2 * C2P);
  for (int c = threadIdx.x; c < C2P; c += 128) {
    float mx = -1e30f, sum = 0.f;
    for (int k = 0; k < KSAMP; ++k) {
      float v = ldsH2[k * C2P + c];
      mx = fmaxf(mx, v); sum += v;
    }
    o[c] = mx; o[C2P + c] = sum * inv;
  }
}

// -------- sa3: [128x544] @ [544x1024] per batch, relu, atomic row-max pool --------
__global__ __launch_bounds__(128)
void sa3_gemm_max_kernel(const uint16_t* __restrict__ A, const uint32_t* __restrict__ Wf,
                         const float* __restrict__ bias, unsigned* __restrict__ outMax) {
  int b = blockIdx.x, mblk = blockIdx.y, nblk = blockIdx.z;
  int wave = threadIdx.x >> 5, lane = threadIdx.x & 31;
  const uint16_t* Ab = A + (size_t)b * 128 * 544;     // fragment-major, MT=8, KS=17
  for (int t = wave; t < 8; t += 4) {
    int mt = t & 1, nt = t >> 1;
    v8f acc = {};
    for (int ks = 0; ks < 544 / 32; ++ks) {
      v16bf af = load_frag_sw(Ab, 17, mblk * 2 + mt, ks, lane);
      v16bf bf = load_frag_sw((const uint16_t*)Wf, 17, nblk * 4 + nt, ks, lane);
      acc = wmma_bf16(af, bf, acc);
    }
    int n = (nblk * 4 + nt) * 16 + (lane & 15);
    float bi = bias[n];
#pragma unroll
    for (int r = 0; r < 8; ++r) {
      float v = acc[r] + bi;
      v = v > 0.f ? v : 0.f;           // relu >= 0 => float bits are max-orderable as uint
      atomicMax(&outMax[(size_t)b * 1024 + n], __float_as_uint(v));
    }
  }
}

// -------- FC head: M=32 WMMA GEMM, optional leaky relu, f32 and/or swizzled bf16 out --------
template<int KDIMP, int NP, int ACT>
__global__ __launch_bounds__(128)
void fc_kernel(const uint16_t* __restrict__ A, const uint32_t* __restrict__ Wf,
               const float* __restrict__ bias, float* __restrict__ outF,
               uint16_t* __restrict__ outBf) {
  int wave = threadIdx.x >> 5, lane = threadIdx.x & 31;
  constexpr int NT = NP / 16, KS = KDIMP / 32, KS2 = NP / 32;
  for (int t = wave; t < 2 * NT; t += 4) {
    int mt = t & 1, nt = t >> 1;
    v8f acc = {};
    for (int ks = 0; ks < KS; ++ks) {
      v16bf af = load_frag_sw(A, KS, mt, ks, lane);
      v16bf bf = load_frag_sw((const uint16_t*)Wf, KS, nt, ks, lane);
      acc = wmma_bf16(af, bf, acc);
    }
    int n = nt * 16 + (lane & 15);
    float bi = bias[n];
    int mofs = mt * 16 + ((lane >> 4) << 3);
#pragma unroll
    for (int r = 0; r < 8; ++r) {
      float v = acc[r] + bi;
      if (ACT == 2) v = v > 0.f ? v : 0.2f * v;
      if (outF)  outF [(mofs + r) * NP + n] = v;
      if (outBf) outBf[frag_half_index(mofs + r, n, KS2)] = f2bf(v);
    }
  }
}

// ============================== host launcher ==============================
struct LP { const float *w, *b, *g, *be; };

extern "C" void kernel_launch(void* const* d_in, const int* in_sizes, int n_in,
                              void* d_out, int out_size, void* d_ws, size_t ws_size,
                              hipStream_t stream) {
  (void)n_in; (void)out_size; (void)ws_size;
  const int B = 32;
  // ---- input mapping: insertion order (xyz first) vs pytree sorted-key order ----
  LP L[10]; const float* xyz;
  if (in_sizes[0] == 32 * 3 * 2048) {
    xyz = (const float*)d_in[0];
    int p = 1;
    for (int i = 0; i < 9; ++i) {  // re0,re1,sa10,sa11,sa20,sa21,sa3,fc1,fc2 : {w,b,g,be}
      L[i].w = (const float*)d_in[p]; L[i].b = (const float*)d_in[p + 1];
      L[i].g = (const float*)d_in[p + 2]; L[i].be = (const float*)d_in[p + 3]; p += 4;
    }
    L[9].w = (const float*)d_in[p]; L[9].b = (const float*)d_in[p + 1];
    L[9].g = nullptr; L[9].be = nullptr;
  } else {
    // sorted keys: fc1{b,be,g,w},fc2,fc3{b,w},re[2],sa1[2],sa2[2],sa3, then xyz
    auto S4 = [&](int base, LP& l) {
      l.b = (const float*)d_in[base]; l.be = (const float*)d_in[base + 1];
      l.g = (const float*)d_in[base + 2]; l.w = (const float*)d_in[base + 3];
    };
    S4(0, L[7]); S4(4, L[8]);
    L[9].b = (const float*)d_in[8]; L[9].w = (const float*)d_in[9]; L[9].g = nullptr; L[9].be = nullptr;
    S4(10, L[0]); S4(14, L[1]); S4(18, L[2]); S4(22, L[3]);
    S4(26, L[4]); S4(30, L[5]); S4(34, L[6]);
    xyz = (const float*)d_in[38];
  }

  // ---- workspace layout ----
  char* ws = (char*)d_ws;
  size_t off = 0;
  auto alloc = [&](size_t bytes) { size_t o = off; off += (bytes + 255) & ~(size_t)255; return o; };
  float*    pts    = (float*)(ws + alloc((size_t)B * 2048 * 3 * 4));
  int*      gidxRe = (int*)  (ws + alloc((size_t)B * 2048 * 32 * 4));
  float*    feats  = (float*)(ws + alloc((size_t)B * 2048 * 64 * 4));
  int*      cent1  = (int*)  (ws + alloc((size_t)B * 512 * 4));
  float*    l1xyz  = (float*)(ws + alloc((size_t)B * 512 * 3 * 4));
  int*      gidx1  = (int*)  (ws + alloc((size_t)B * 512 * 32 * 4));
  float*    l1     = (float*)(ws + alloc((size_t)B * 512 * 128 * 4));
  int*      cent2  = (int*)  (ws + alloc((size_t)B * 128 * 4));
  float*    l2xyz  = (float*)(ws + alloc((size_t)B * 128 * 3 * 4));
  int*      gidx2  = (int*)  (ws + alloc((size_t)B * 128 * 20 * 4));
  float*    l2     = (float*)(ws + alloc((size_t)B * 128 * 512 * 4));
  uint16_t* A3     = (uint16_t*)(ws + alloc((size_t)B * 128 * 544 * 2));
  unsigned* x3max  = (unsigned*)(ws + alloc((size_t)B * 1024 * 4));
  uint16_t* x3bf   = (uint16_t*)(ws + alloc((size_t)B * 1024 * 2));
  uint16_t* x1bf   = (uint16_t*)(ws + alloc((size_t)B * 512 * 2));
  uint16_t* x2bf   = (uint16_t*)(ws + alloc((size_t)B * 256 * 2));
  float*    fc3out = (float*)(ws + alloc((size_t)B * 48 * 4));

  struct WInfo { int Cin, Cout, CinP, CoutP; };
  const WInfo wi[10] = {
    {6, 32, 32, 32},   {32, 32, 32, 32},
    {70, 64, 96, 64},  {64, 64, 64, 64},
    {134, 128, 160, 128}, {128, 256, 128, 256},
    {518, 1024, 544, 1024},
    {1024, 512, 1024, 512}, {512, 256, 512, 256}, {256, 40, 256, 48}
  };
  uint32_t* Wf[10]; float* bs[10];
  for (int i = 0; i < 10; ++i) {
    Wf[i] = (uint32_t*)(ws + alloc((size_t)(wi[i].CinP / 2) * wi[i].CoutP * 4));
    bs[i] = (float*)  (ws + alloc((size_t)wi[i].CoutP * 4));
  }

  // ---- pipeline ----
  transpose_kernel<<<(B * 2048 * 3 + 255) / 256, 256, 0, stream>>>(xyz, pts, 2048);
  for (int i = 0; i < 10; ++i) {
    int tot = (wi[i].CinP / 2) * wi[i].CoutP + wi[i].CoutP;
    prep_w_kernel<<<(tot + 255) / 256, 256, 0, stream>>>(L[i].w, L[i].b, L[i].g, L[i].be,
        Wf[i], bs[i], wi[i].Cin, wi[i].Cout, wi[i].CinP, wi[i].CoutP);
  }
  // re stage (all points as centers)
  ball_query_kernel<<<(B * 2048) / 4, 128, 0, stream>>>(pts, nullptr, gidxRe, 2048, 2048, B * 2048, 0.04f, 32);
  group_mlp2_pool_kernel<32, 32, 0, 32, 32, 32, false><<<B * 2048, 128, 0, stream>>>(
      pts, nullptr, nullptr, gidxRe, Wf[0], bs[0], Wf[1], bs[1], feats, 2048, 2048);
  // sa1
  fps_kernel<2048><<<B, 256, 0, stream>>>(pts, cent1, 2048, 512);
  gather_xyz_kernel<<<(B * 512 * 3 + 255) / 256, 256, 0, stream>>>(pts, cent1, l1xyz, 2048, 512);
  ball_query_kernel<<<(B * 512) / 4, 128, 0, stream>>>(pts, cent1, gidx1, 2048, 512, B * 512, 0.04f, 32);
  group_mlp2_pool_kernel<32, 32, 64, 96, 64, 64, true><<<B * 512, 128, 0, stream>>>(
      pts, feats, cent1, gidx1, Wf[2], bs[2], Wf[3], bs[3], l1, 2048, 512);
  // sa2
  fps_kernel<512><<<B, 256, 0, stream>>>(l1xyz, cent2, 512, 128);
  gather_xyz_kernel<<<(B * 128 * 3 + 255) / 256, 256, 0, stream>>>(l1xyz, cent2, l2xyz, 512, 128);
  ball_query_kernel<<<(B * 128) / 4, 128, 0, stream>>>(l1xyz, cent2, gidx2, 512, 128, B * 128, 0.16f, 20);
  group_mlp2_pool_kernel<20, 32, 128, 160, 128, 256, true><<<B * 128, 128, 0, stream>>>(
      l1xyz, l1, cent2, gidx2, Wf[4], bs[4], Wf[5], bs[5], l2, 512, 128);
  // sa3
  build_a3_kernel<<<(B * 128 * 544 + 255) / 256, 256, 0, stream>>>(l2, l2xyz, A3);
  zero_u32_kernel<<<(B * 1024 + 255) / 256, 256, 0, stream>>>(x3max, B * 1024);
  sa3_gemm_max_kernel<<<dim3(B, 4, 16), 128, 0, stream>>>(A3, Wf[6], bs[6], x3max);
  u2bf_kernel<<<(B * 1024 + 255) / 256, 256, 0, stream>>>(x3max, x3bf, B * 1024);
  // FC head
  fc_kernel<1024, 512, 2><<<1, 128, 0, stream>>>(x3bf, Wf[7], bs[7], (float*)nullptr, x1bf);
  fc_kernel<512, 256, 2><<<1, 128, 0, stream>>>(x1bf, Wf[8], bs[8], (float*)nullptr, x2bf);
  fc_kernel<256, 48, 0><<<1, 128, 0, stream>>>(x2bf, Wf[9], bs[9], fc3out, (uint16_t*)nullptr);
  finalize_kernel<<<(1283 + 255) / 256, 256, 0, stream>>>(fc3out, (float*)d_out);
}
